// NDT2Model_7627861917863
// MI455X (gfx1250) — matrix operand order repaired and checked by reading, not compiled, not run on hardware
//
#include <hip/hip_runtime.h>
#include <math.h>

// ---------------- model constants ----------------
#define BB     8
#define SLEN   1024
#define CTX    2
#define LTOT   1026          // S + C
#define DMODEL 512
#define NHEAD  8
#define HD     64
#define FFDIM  2048
#define DEPTH  6
#define TRIPLE 1536          // 3*D
#define MROWS  8208          // B * LTOT  (= 513 * 16)
#define LPAD   1056          // padded key stride for vT (mult of 16, covers last tile overread)
#define NEGBIG (-1.0e30f)

// ---------------- WMMA types ----------------
typedef __bf16 bf16_t;
typedef bf16_t v16bf __attribute__((ext_vector_type(16)));
typedef float  v8f   __attribute__((ext_vector_type(8)));
typedef unsigned int u32x4 __attribute__((ext_vector_type(4)));

union Frag {
    v16bf v;
    unsigned short u[16];
    u32x4 q[2];
};

__device__ inline unsigned short f2bf(float f) {
    unsigned int u = __float_as_uint(f);
    unsigned int r = u + 0x7FFFu + ((u >> 16) & 1u);   // round-to-nearest-even
    return (unsigned short)(r >> 16);
}

__device__ inline v8f zero8() {
    v8f z;
#pragma unroll
    for (int i = 0; i < 8; i++) z[i] = 0.0f;
    return z;
}

__device__ inline v8f wmma_bf16(v16bf a, v16bf b, v8f c) {
    return __builtin_amdgcn_wmma_f32_16x16x32_bf16(false, a, false, b, (short)0, c,
                                                   false, false);
}

// A-matrix 16x32 bf16 fragment: lane (m = lane&15) holds row m.
// elements 0..7  -> K = k + hi*8 + {0..7}
// elements 8..15 -> K = k + 16 + hi*8 + {0..7}
__device__ inline v16bf load_fragA(const unsigned short* base, int stride, int row,
                                   int k, int hi) {
    Frag f;
    const unsigned short* p = base + (size_t)row * stride + k + hi * 8;
    f.q[0] = *(const u32x4*)(p);
    f.q[1] = *(const u32x4*)(p + 16);
    return f.v;
}

// B-matrix 32x16 bf16 fragment: lane (n = lane&15) holds column n.
// elements 0..7  -> K = k + hi*8 + {0..7}; elements 8..15 -> K = k + 16 + hi*8 + {0..7}
__device__ inline v16bf load_fragB(const unsigned short* base, int stride, int row,
                                   int k, int hi) {
    Frag f;
    const unsigned short* p = base + (size_t)row * stride + k + hi * 8;
    f.q[0] = *(const u32x4*)(p);
    f.q[1] = *(const u32x4*)(p + 16);
    return f.v;
}

// A fragment from a 16x32 bf16 tile in LDS (row-major, 32 elems/row)
__device__ inline v16bf load_fragA_lds(const unsigned short* pl, int row, int hi) {
    Frag f;
    const unsigned short* p = pl + row * 32 + hi * 8;
    f.q[0] = *(const u32x4*)(p);
    f.q[1] = *(const u32x4*)(p + 16);
    return f.v;
}

__device__ inline void dswait() {
#if __has_builtin(__builtin_amdgcn_s_wait_dscnt)
    __builtin_amdgcn_s_wait_dscnt(0);
#else
    asm volatile("s_wait_dscnt 0" ::: "memory");
#endif
}

// ---------------- elementwise kernels ----------------

__global__ __launch_bounds__(256) void cvt_bf16_kernel(const float* __restrict__ s,
                                                       unsigned short* __restrict__ d,
                                                       int n) {
    int i = blockIdx.x * 256 + threadIdx.x;
    if (i < n) d[i] = f2bf(s[i]);
}

__global__ __launch_bounds__(256) void embed_kernel(
    const float* __restrict__ src, const float* __restrict__ ctx,
    const int* __restrict__ times, const int* __restrict__ spaces,
    const float* __restrict__ temb, const float* __restrict__ semb,
    float* __restrict__ xf, unsigned short* __restrict__ xh) {
    int idx = blockIdx.x * 256 + threadIdx.x;
    if (idx >= MROWS * DMODEL) return;
    int d = idx & (DMODEL - 1);
    int row = idx >> 9;
    int b = row / LTOT;
    int l = row - b * LTOT;
    float v;
    if (l < SLEN) {
        int t = times[b * SLEN + l];
        int sp = spaces[b * SLEN + l];
        v = src[((size_t)b * SLEN + l) * DMODEL + d] + temb[(size_t)t * DMODEL + d] +
            semb[(size_t)sp * DMODEL + d];
    } else {
        v = ctx[((size_t)b * CTX + (l - SLEN)) * DMODEL + d];
    }
    xf[idx] = v;
    xh[idx] = f2bf(v);
}

// transpose V out of the packed qkv buffer:  vT[(b,h,d), l]  (key dim innermost, stride LPAD)
__global__ __launch_bounds__(256) void vtrans_kernel(
    const unsigned short* __restrict__ qkvh, unsigned short* __restrict__ vT) {
    int idx = blockIdx.x * 256 + threadIdx.x;
    if (idx >= BB * NHEAD * HD * LTOT) return;
    int l = idx % LTOT;
    int rem = idx / LTOT;   // (b*NHEAD + h)*HD + d
    int d = rem & (HD - 1);
    int bh = rem >> 6;
    int h = bh & (NHEAD - 1);
    int b = bh >> 3;
    vT[(size_t)rem * LPAD + l] =
        qkvh[((size_t)b * LTOT + l) * TRIPLE + 2 * DMODEL + h * HD + d];
}

__global__ __launch_bounds__(256) void copyout_kernel(const float* __restrict__ xf,
                                                      float* __restrict__ out) {
    int idx = blockIdx.x * 256 + threadIdx.x;
    if (idx >= BB * SLEN * DMODEL) return;
    int d = idx & (DMODEL - 1);
    int rs = idx >> 9;
    int b = rs >> 10;
    int s = rs & (SLEN - 1);
    out[idx] = xf[((size_t)b * LTOT + s) * DMODEL + d];
}

// x = LN(x + y) * w + b   (one block per row, D = 512, 256 threads x 2 elems)
__global__ __launch_bounds__(256) void add_ln_kernel(
    const float* __restrict__ x, const float* __restrict__ y,
    const float* __restrict__ w, const float* __restrict__ bp,
    float* __restrict__ xo, unsigned short* __restrict__ xh) {
    int row = blockIdx.x;
    int tid = threadIdx.x;
    const float* xr = x + (size_t)row * DMODEL;
    const float* yr = y + (size_t)row * DMODEL;
    float v0 = xr[tid] + yr[tid];
    float v1 = xr[tid + 256] + yr[tid + 256];
    float s = v0 + v1, sq = v0 * v0 + v1 * v1;
#pragma unroll
    for (int off = 16; off; off >>= 1) {
        s += __shfl_xor(s, off, 32);
        sq += __shfl_xor(sq, off, 32);
    }
    __shared__ float ss[8], sqq[8];
    int wave = tid >> 5, lane = tid & 31;
    if (lane == 0) { ss[wave] = s; sqq[wave] = sq; }
    __syncthreads();
    if (tid == 0) {
        float a = 0.f, c = 0.f;
#pragma unroll
        for (int i = 0; i < 8; i++) { a += ss[i]; c += sqq[i]; }
        ss[0] = a; sqq[0] = c;
    }
    __syncthreads();
    float mean = ss[0] * (1.0f / DMODEL);
    float var = sqq[0] * (1.0f / DMODEL) - mean * mean;
    float inv = rsqrtf(var + 1e-5f);
    float o0 = (v0 - mean) * inv * w[tid] + bp[tid];
    float o1 = (v1 - mean) * inv * w[tid + 256] + bp[tid + 256];
    float* xout = xo + (size_t)row * DMODEL;
    unsigned short* xhh = xh + (size_t)row * DMODEL;
    xout[tid] = o0;
    xout[tid + 256] = o1;
    xhh[tid] = f2bf(o0);
    xhh[tid + 256] = f2bf(o1);
}

// ---------------- WMMA GEMM:  Y[M,N] = A[M,K](bf16) @ W[N,K]^T(bf16) + bias ----------------
// 256 threads = 8 waves arranged 2(M) x 4(N); each wave computes a 16x64 strip:
// one A fragment reused across 4 B fragments / 4 WMMAs per K-step.
template <bool GELU, bool F32OUT, bool BF16OUT>
__global__ __launch_bounds__(256) void gemm_kernel(
    const unsigned short* __restrict__ A, const unsigned short* __restrict__ W,
    const float* __restrict__ bias, float* __restrict__ Yf,
    unsigned short* __restrict__ Yh, int M, int N, int K) {
    int tid = threadIdx.x;
    int wave = tid >> 5, lane = tid & 31;
    int hi = lane >> 4, li = lane & 15;
    int mtile = blockIdx.x * 2 + (wave >> 2);
    if (mtile * 16 >= M) return;
    int n0 = blockIdx.y * 256 + (wave & 3) * 64;
    int rowA = mtile * 16 + li;
    int rowW = n0 + li;
    v8f acc[4];
#pragma unroll
    for (int j = 0; j < 4; j++) acc[j] = zero8();
#pragma unroll 2
    for (int k = 0; k < K; k += 32) {
        v16bf a = load_fragA(A, K, rowA, k, hi);
        v16bf b0 = load_fragB(W, K, rowW, k, hi);
        v16bf b1 = load_fragB(W, K, rowW + 16, k, hi);
        v16bf b2 = load_fragB(W, K, rowW + 32, k, hi);
        v16bf b3 = load_fragB(W, K, rowW + 48, k, hi);
        acc[0] = wmma_bf16(a, b0, acc[0]);
        acc[1] = wmma_bf16(a, b1, acc[1]);
        acc[2] = wmma_bf16(a, b2, acc[2]);
        acc[3] = wmma_bf16(a, b3, acc[3]);
    }
#pragma unroll
    for (int j = 0; j < 4; j++) {
        int col = n0 + j * 16 + li;
        float bv = bias[col];
#pragma unroll
        for (int r = 0; r < 8; r++) {
            int row = mtile * 16 + r + hi * 8;
            float v = acc[j][r] + bv;
            if (GELU) v = 0.5f * v * (1.0f + erff(v * 0.70710678118654752f));
            size_t off = (size_t)row * N + col;
            if (F32OUT) Yf[off] = v;
            if (BF16OUT) Yh[off] = f2bf(v);
        }
    }
}

// ---------------- flash attention: one wave per 16-query tile per (b,h) ----------------
// qkv: bf16 (B, LTOT, 3*D) for Q,K; vT: bf16 (B,H,HD, LPAD); out: bf16 (B, LTOT, D)
__global__ __launch_bounds__(128) void attn_kernel(
    const unsigned short* __restrict__ qkv, const unsigned short* __restrict__ vT,
    const int* __restrict__ times, const unsigned char* __restrict__ padm,
    unsigned short* __restrict__ aout) {
    __shared__ unsigned short plds[4 * 16 * 32];
    int tid = threadIdx.x;
    int wave = tid >> 5, lane = tid & 31;
    int hi = lane >> 4, li = lane & 15;
    int qtile = blockIdx.x * 4 + wave;
    int q0 = qtile * 16;
    if (q0 >= LTOT) return;
    int h = blockIdx.y, b = blockIdx.z;

    const unsigned short* base = qkv + (size_t)b * LTOT * TRIPLE + h * HD;
    const unsigned short* qb = base;
    const unsigned short* kb = base + DMODEL;
    const unsigned short* vt = vT + ((size_t)(b * NHEAD + h) * HD) * LPAD;
    unsigned short* pl = plds + wave * 512;

    int rowq = q0 + li;
    if (rowq > LTOT - 1) rowq = LTOT - 1;
    v16bf qa0 = load_fragA(qb, TRIPLE, rowq, 0, hi);
    v16bf qa1 = load_fragA(qb, TRIPLE, rowq, 32, hi);

    int tq[8];
#pragma unroll
    for (int r = 0; r < 8; r++) {
        int row = q0 + r + hi * 8;
        tq[r] = (row < SLEN) ? times[b * SLEN + row] : 0;
    }

    float mrow[8], lrow[8];
    v8f o0 = zero8(), o1 = zero8(), o2 = zero8(), o3 = zero8();
#pragma unroll
    for (int r = 0; r < 8; r++) { mrow[r] = NEGBIG; lrow[r] = 0.0f; }
    const float scale = 0.125f;  // 1/sqrt(64)

    for (int kt = 0; kt < LTOT; kt += 32) {
        int kk0 = kt + li, kk1 = kt + 16 + li;
        int rk0 = kk0 < LTOT ? kk0 : LTOT - 1;
        int rk1 = kk1 < LTOT ? kk1 : LTOT - 1;
        v16bf b00 = load_fragB(kb, TRIPLE, rk0, 0, hi);
        v16bf b01 = load_fragB(kb, TRIPLE, rk0, 32, hi);
        v16bf b10 = load_fragB(kb, TRIPLE, rk1, 0, hi);
        v16bf b11 = load_fragB(kb, TRIPLE, rk1, 32, hi);
        v8f s0 = zero8(), s1 = zero8();
        s0 = wmma_bf16(qa0, b00, s0);
        s0 = wmma_bf16(qa1, b01, s0);
        s1 = wmma_bf16(qa0, b10, s1);
        s1 = wmma_bf16(qa1, b11, s1);

        bool v0ok = kk0 < LTOT, v1ok = kk1 < LTOT;
        bool pm0 = v0ok && (padm[b * LTOT + kk0] != 0);
        bool pm1 = v1ok && (padm[b * LTOT + kk1] != 0);
        bool ctx0 = kk0 >= SLEN, ctx1 = kk1 >= SLEN;
        int tk0 = (v0ok && !ctx0) ? times[b * SLEN + kk0] : 0;
        int tk1 = (v1ok && !ctx1) ? times[b * SLEN + kk1] : 0;

#pragma unroll
        for (int r = 0; r < 8; r++) {
            int row = q0 + r + hi * 8;
            float sv0 = s0[r] * scale, sv1 = s1[r] * scale;
            if (!v0ok || pm0) sv0 = NEGBIG;
            else if (!ctx0) {
                if (row >= SLEN || tq[r] < tk0) sv0 = NEGBIG;
            }
            if (!v1ok || pm1) sv1 = NEGBIG;
            else if (!ctx1) {
                if (row >= SLEN || tq[r] < tk1) sv1 = NEGBIG;
            }
            float mv = fmaxf(sv0, sv1);
#pragma unroll
            for (int off = 8; off; off >>= 1) mv = fmaxf(mv, __shfl_xor(mv, off, 16));
            float nm = fmaxf(mrow[r], mv);
            float alpha = __expf(mrow[r] - nm);
            float p0 = __expf(sv0 - nm), p1 = __expf(sv1 - nm);
            float ps = p0 + p1;
#pragma unroll
            for (int off = 8; off; off >>= 1) ps += __shfl_xor(ps, off, 16);
            lrow[r] = lrow[r] * alpha + ps;
            mrow[r] = nm;
            o0[r] = o0[r] * alpha;
            o1[r] = o1[r] * alpha;
            o2[r] = o2[r] * alpha;
            o3[r] = o3[r] * alpha;
            int prow = r + hi * 8;
            pl[prow * 32 + li] = f2bf(p0);
            pl[prow * 32 + 16 + li] = f2bf(p1);
        }
        dswait();  // wave-private LDS tile visible before re-read

        v16bf pa = load_fragA_lds(pl, li, hi);
        // V B-fragments: contiguous along padded key dim of vT (row = head-dim col)
        v16bf vb0 = load_fragB(vt, LPAD, 0 * 16 + li, kt, hi);
        v16bf vb1 = load_fragB(vt, LPAD, 1 * 16 + li, kt, hi);
        v16bf vb2 = load_fragB(vt, LPAD, 2 * 16 + li, kt, hi);
        v16bf vb3 = load_fragB(vt, LPAD, 3 * 16 + li, kt, hi);
        o0 = wmma_bf16(pa, vb0, o0);
        o1 = wmma_bf16(pa, vb1, o1);
        o2 = wmma_bf16(pa, vb2, o2);
        o3 = wmma_bf16(pa, vb3, o3);
    }

#pragma unroll
    for (int r = 0; r < 8; r++) {
        int row = q0 + r + hi * 8;
        if (row >= LTOT) continue;
        float inv = 1.0f / lrow[r];
        size_t ob = ((size_t)b * LTOT + row) * DMODEL + h * HD + li;
        aout[ob] = f2bf(o0[r] * inv);
        aout[ob + 16] = f2bf(o1[r] * inv);
        aout[ob + 32] = f2bf(o2[r] * inv);
        aout[ob + 48] = f2bf(o3[r] * inv);
    }
}

// ---------------- host orchestration ----------------
extern "C" void kernel_launch(void* const* d_in, const int* in_sizes, int n_in,
                              void* d_out, int out_size, void* d_ws, size_t ws_size,
                              hipStream_t stream) {
    const float* src = (const float*)d_in[0];
    const float* ctx = (const float*)d_in[1];
    const int* times = (const int*)d_in[2];
    const int* spaces = (const int*)d_in[3];
    const unsigned char* padm = (const unsigned char*)d_in[4];
    const float* temb = (const float*)d_in[5];
    const float* semb = (const float*)d_in[6];
    const float* Wqkv = (const float*)d_in[7];
    const float* bqkv = (const float*)d_in[8];
    const float* Wo = (const float*)d_in[9];
    const float* bo = (const float*)d_in[10];
    const float* ln1w = (const float*)d_in[11];
    const float* ln1b = (const float*)d_in[12];
    const float* W1 = (const float*)d_in[13];
    const float* b1 = (const float*)d_in[14];
    const float* W2 = (const float*)d_in[15];
    const float* b2 = (const float*)d_in[16];
    const float* ln2w = (const float*)d_in[17];
    const float* ln2b = (const float*)d_in[18];

    char* ws = (char*)d_ws;
    size_t off = 0;
    auto alloc = [&](size_t bytes) -> void* {
        void* p = ws + off;
        off += (bytes + 255) & ~(size_t)255;
        return p;
    };
    float* xf = (float*)alloc((size_t)MROWS * DMODEL * 4);
    unsigned short* xh = (unsigned short*)alloc((size_t)MROWS * DMODEL * 2);
    unsigned short* qkvh = (unsigned short*)alloc((size_t)MROWS * TRIPLE * 2);
    unsigned short* vT = (unsigned short*)alloc((size_t)BB * NHEAD * HD * LPAD * 2);
    float* abuf = (float*)alloc((size_t)MROWS * DMODEL * 4);
    unsigned short* hbuf = (unsigned short*)alloc((size_t)MROWS * FFDIM * 2);
    unsigned short* wq_h = (unsigned short*)alloc((size_t)TRIPLE * DMODEL * 2);
    unsigned short* wo_h = (unsigned short*)alloc((size_t)DMODEL * DMODEL * 2);
    unsigned short* w1_h = (unsigned short*)alloc((size_t)FFDIM * DMODEL * 2);
    unsigned short* w2_h = (unsigned short*)alloc((size_t)DMODEL * FFDIM * 2);

    {
        int n = MROWS * DMODEL;
        embed_kernel<<<(n + 255) / 256, 256, 0, stream>>>(src, ctx, times, spaces, temb,
                                                          semb, xf, xh);
    }

    const int nWqkv = TRIPLE * DMODEL;   // 786432
    const int nWo = DMODEL * DMODEL;     // 262144
    const int nW1 = FFDIM * DMODEL;      // 1048576
    const int nW2 = DMODEL * FFDIM;      // 1048576

    for (int i = 0; i < DEPTH; i++) {
        cvt_bf16_kernel<<<(nWqkv + 255) / 256, 256, 0, stream>>>(
            Wqkv + (size_t)i * nWqkv, wq_h, nWqkv);
        cvt_bf16_kernel<<<(nWo + 255) / 256, 256, 0, stream>>>(Wo + (size_t)i * nWo,
                                                               wo_h, nWo);
        cvt_bf16_kernel<<<(nW1 + 255) / 256, 256, 0, stream>>>(W1 + (size_t)i * nW1,
                                                               w1_h, nW1);
        cvt_bf16_kernel<<<(nW2 + 255) / 256, 256, 0, stream>>>(W2 + (size_t)i * nW2,
                                                               w2_h, nW2);

        // qkv = x @ Wqkv^T + bqkv   -> bf16
        gemm_kernel<false, false, true><<<dim3(257, TRIPLE / 256), 256, 0, stream>>>(
            xh, wq_h, bqkv + (size_t)i * TRIPLE, nullptr, qkvh, MROWS, TRIPLE, DMODEL);

        // transpose V for contiguous B-fragments in attention
        {
            int n = BB * NHEAD * HD * LTOT;
            vtrans_kernel<<<(n + 255) / 256, 256, 0, stream>>>(qkvh, vT);
        }

        // attention -> bf16 into hbuf (first 8.4MB of it)
        attn_kernel<<<dim3(17, NHEAD, BB), 128, 0, stream>>>(qkvh, vT, times, padm,
                                                             hbuf);

        // a = attn @ Wo^T + bo -> fp32
        gemm_kernel<false, true, false><<<dim3(257, DMODEL / 256), 256, 0, stream>>>(
            hbuf, wo_h, bo + (size_t)i * DMODEL, abuf, nullptr, MROWS, DMODEL, DMODEL);

        // x = LN(x + a)
        add_ln_kernel<<<MROWS, 256, 0, stream>>>(xf, abuf, ln1w + (size_t)i * DMODEL,
                                                 ln1b + (size_t)i * DMODEL, xf, xh);

        // h = gelu(x @ W1^T + b1) -> bf16
        gemm_kernel<true, false, true><<<dim3(257, FFDIM / 256), 256, 0, stream>>>(
            xh, w1_h, b1 + (size_t)i * FFDIM, nullptr, hbuf, MROWS, FFDIM, DMODEL);

        // y = h @ W2^T + b2 -> fp32
        gemm_kernel<false, true, false><<<dim3(257, DMODEL / 256), 256, 0, stream>>>(
            hbuf, w2_h, b2 + (size_t)i * DMODEL, abuf, nullptr, MROWS, DMODEL, FFDIM);

        // x = LN(x + y)
        add_ln_kernel<<<MROWS, 256, 0, stream>>>(xf, abuf, ln2w + (size_t)i * DMODEL,
                                                 ln2b + (size_t)i * DMODEL, xf, xh);
    }

    {
        int n = BB * SLEN * DMODEL;
        copyout_kernel<<<(n + 255) / 256, 256, 0, stream>>>(xf, (float*)d_out);
    }
}
